// DebugQwen3Attention_70360154243469
// MI455X (gfx1250) — matrix operand order repaired and checked by reading, not compiled, not run on hardware
//
#include <hip/hip_runtime.h>
#include <hip/hip_bf16.h>
#include <math.h>

// ---------------- types ----------------
typedef __attribute__((ext_vector_type(16))) __bf16 bf16x16;
typedef __attribute__((ext_vector_type(8)))  __bf16 bf16x8;
typedef __attribute__((ext_vector_type(8)))  float  f32x8;

#define S_LEN  2048
#define DMODEL 2048
#define NHEADS 16
#define NKV    8
#define HDIM   128
#define NREP   2
#define MASKV  (-3.4028234663852886e38f)
#define BN     64      // block N tile
#define BK     32      // K step (one WMMA depth)

__device__ __forceinline__ __bf16 f2bf(float x) {
    unsigned u = __builtin_bit_cast(unsigned, x);
    unsigned r = u + 0x7FFFu + ((u >> 16) & 1u);     // round-to-nearest-even
    unsigned short h = (unsigned short)(r >> 16);
    return __builtin_bit_cast(__bf16, h);
}

// ---------------- CDNA5 async copy helpers ----------------
// LDS byte address = low 32 bits of the generic pointer to shared memory
// (flat LDS aperture keeps the wave-relative offset in addr[31:0]).
__device__ __forceinline__ unsigned lds_addr32(const void* p) {
    return (unsigned)(unsigned long long)p;
}

// One per-thread 16B element of a 64(N) x 32(K) bf16 tile: 256 threads cover 4KB.
// Emits GLOBAL_LOAD_ASYNC_TO_LDS_B128 (tracked by ASYNCcnt, not LOADcnt).
__device__ __forceinline__ void stage_b_async(const __bf16* B, int ldb, int col0, int k0,
                                              __bf16* dstLds, int tid) {
    int n = tid >> 2;          // 0..63 : row of tile (output col)
    int c = tid & 3;           // 0..3  : 8-element chunk along K
    const __bf16* gp = B + (size_t)(col0 + n) * ldb + (size_t)(k0 + c * 8);
    unsigned long long ga = (unsigned long long)gp;
    unsigned la = lds_addr32(dstLds + n * BK + c * 8);
    asm volatile("global_load_async_to_lds_b128 %0, %1, off"
                 :: "v"(la), "v"(ga) : "memory");
}

__device__ __forceinline__ void wait_async0() {
    asm volatile("s_wait_asynccnt 0x0" ::: "memory");
}

// ---------------- WMMA fragment loads ----------------
// A fragment (16x32 bf16, M x K): lane r=l&15 -> row r; half=l>>4:
// frag[0..7]=K[half*8 .. half*8+7], frag[8..15]=K[16+half*8 ..]
__device__ __forceinline__ bf16x16 load_a_bf16(const __bf16* base, size_t ld, int k0, int lane) {
    int r = lane & 15, half = (lane >> 4) & 1;
    const __bf16* p = base + (size_t)r * ld + (size_t)(k0 + half * 8);
    bf16x8 lo = *(const bf16x8*)p;
    bf16x8 hi = *(const bf16x8*)(p + 16);
    bf16x16 f;
#pragma unroll
    for (int i = 0; i < 8; ++i) { f[i] = lo[i]; f[i + 8] = hi[i]; }
    return f;
}

// Same A fragment but from fp32 memory (converted on the fly)
__device__ __forceinline__ bf16x16 load_a_f32(const float* base, size_t ld, int k0, int lane) {
    int r = lane & 15, half = (lane >> 4) & 1;
    const float* p = base + (size_t)r * ld + (size_t)(k0 + half * 8);
    f32x8 lo = *(const f32x8*)p;
    f32x8 hi = *(const f32x8*)(p + 16);
    bf16x16 f;
#pragma unroll
    for (int i = 0; i < 8; ++i) { f[i] = f2bf(lo[i]); f[i + 8] = f2bf(hi[i]); }
    return f;
}

// B fragment (32x16, K x N) from an LDS tile stored [n][k] (BK=32 per row):
// lane n=l&15 -> col n; half=l>>4: frag[0..7]=K[half*8 ..], frag[8..15]=K[16+half*8 ..]  (two 16B ds reads)
__device__ __forceinline__ bf16x16 load_b_lds(const __bf16* tile16 /*base of 16-row group*/,
                                              int lane) {
    int n = lane & 15, half = (lane >> 4) & 1;
    const __bf16* q = tile16 + n * BK + half * 8;
    bf16x8 lo = *(const bf16x8*)q;
    bf16x8 hi = *(const bf16x8*)(q + 16);
    bf16x16 f;
#pragma unroll
    for (int i = 0; i < 8; ++i) { f[i] = lo[i]; f[i + 8] = hi[i]; }
    return f;
}

// ---------------- generic GEMM: C = alpha * A(bf16) * B(bf16)^T (+causal) ----------------
// block = 256 threads = 8 waves; block tile 128(M) x 64(N); wave tile 16 x 64.
// B tile double-buffered in LDS via async-to-LDS DMA.
__global__ __launch_bounds__(256) void gemm_bf16_nt(
    const __bf16* __restrict__ A, const __bf16* __restrict__ B, float* __restrict__ C,
    int K, int lda, int ldb, int ldc,
    long long strideA, long long strideB, int repB, long long strideC,
    float alpha, int causal)
{
    int z = blockIdx.z;
    A += (long long)z * strideA;
    B += (long long)(z / repB) * strideB;
    C += (long long)z * strideC;
    int tid  = threadIdx.x;
    int lane = tid & 31;
    int wave = tid >> 5;
    int row0 = blockIdx.x * 128 + wave * 16;
    int col0 = blockIdx.y * BN;
    const __bf16* Arow = A + (size_t)row0 * lda;

    __shared__ __attribute__((aligned(64))) __bf16 bsh[2][BN * BK];

    stage_b_async(B, ldb, col0, 0, &bsh[0][0], tid);
    wait_async0();
    __syncthreads();

    f32x8 acc[4] = {};
    int p = 0;
    for (int k0 = 0; k0 < K; k0 += BK) {
        bool more = (k0 + BK) < K;
        if (more) {
            stage_b_async(B, ldb, col0, k0 + BK, &bsh[p ^ 1][0], tid);
            __builtin_prefetch(Arow + (size_t)(lane & 15) * lda + (size_t)(k0 + BK), 0, 1);
        }
        bf16x16 a  = load_a_bf16(Arow, lda, k0, lane);
        bf16x16 b0 = load_b_lds(&bsh[p][ 0 * BK], lane);
        bf16x16 b1 = load_b_lds(&bsh[p][16 * BK], lane);
        bf16x16 b2 = load_b_lds(&bsh[p][32 * BK], lane);
        bf16x16 b3 = load_b_lds(&bsh[p][48 * BK], lane);
        acc[0] = __builtin_amdgcn_wmma_f32_16x16x32_bf16(false, a, false, b0, (short)0, acc[0], false, false);
        acc[1] = __builtin_amdgcn_wmma_f32_16x16x32_bf16(false, a, false, b1, (short)0, acc[1], false, false);
        acc[2] = __builtin_amdgcn_wmma_f32_16x16x32_bf16(false, a, false, b2, (short)0, acc[2], false, false);
        acc[3] = __builtin_amdgcn_wmma_f32_16x16x32_bf16(false, a, false, b3, (short)0, acc[3], false, false);
        if (more) {
            wait_async0();        // next tile landed in LDS
            __syncthreads();      // all waves done reading buf p, new data visible
            p ^= 1;
        }
    }

    int cr = lane & 15;
    int rbase = row0 + ((lane >> 4) & 1) * 8;
#pragma unroll
    for (int r = 0; r < 8; ++r) {
        int row = rbase + r;
        float* crow = C + (size_t)row * ldc;
#pragma unroll
        for (int t = 0; t < 4; ++t) {
            int col = col0 + t * 16 + cr;
            float v = acc[t][r] * alpha;
            if (causal && col > row) v += MASKV;
            crow[col] = v;
        }
    }
}

// ---------------- GEMM with fp32 A (attn probs), same LDS-async B path ----------------
__global__ __launch_bounds__(256) void gemm_f32a_nt(
    const float* __restrict__ A, const __bf16* __restrict__ B, float* __restrict__ C,
    int K, int lda, int ldb, int ldc,
    long long strideA, long long strideB, int repB, long long strideC)
{
    int z = blockIdx.z;
    A += (long long)z * strideA;
    B += (long long)(z / repB) * strideB;
    C += (long long)z * strideC;
    int tid  = threadIdx.x;
    int lane = tid & 31;
    int wave = tid >> 5;
    int row0 = blockIdx.x * 128 + wave * 16;
    int col0 = blockIdx.y * BN;
    const float* Arow = A + (size_t)row0 * lda;

    __shared__ __attribute__((aligned(64))) __bf16 bsh[2][BN * BK];

    stage_b_async(B, ldb, col0, 0, &bsh[0][0], tid);
    wait_async0();
    __syncthreads();

    f32x8 acc[4] = {};
    int p = 0;
    for (int k0 = 0; k0 < K; k0 += BK) {
        bool more = (k0 + BK) < K;
        if (more) {
            stage_b_async(B, ldb, col0, k0 + BK, &bsh[p ^ 1][0], tid);
            __builtin_prefetch(Arow + (size_t)(lane & 15) * lda + (size_t)(k0 + BK), 0, 1);
        }
        bf16x16 a  = load_a_f32(Arow, lda, k0, lane);
        bf16x16 b0 = load_b_lds(&bsh[p][ 0 * BK], lane);
        bf16x16 b1 = load_b_lds(&bsh[p][16 * BK], lane);
        bf16x16 b2 = load_b_lds(&bsh[p][32 * BK], lane);
        bf16x16 b3 = load_b_lds(&bsh[p][48 * BK], lane);
        acc[0] = __builtin_amdgcn_wmma_f32_16x16x32_bf16(false, a, false, b0, (short)0, acc[0], false, false);
        acc[1] = __builtin_amdgcn_wmma_f32_16x16x32_bf16(false, a, false, b1, (short)0, acc[1], false, false);
        acc[2] = __builtin_amdgcn_wmma_f32_16x16x32_bf16(false, a, false, b2, (short)0, acc[2], false, false);
        acc[3] = __builtin_amdgcn_wmma_f32_16x16x32_bf16(false, a, false, b3, (short)0, acc[3], false, false);
        if (more) {
            wait_async0();
            __syncthreads();
            p ^= 1;
        }
    }

    int cr = lane & 15;
    int rbase = row0 + ((lane >> 4) & 1) * 8;
#pragma unroll
    for (int r = 0; r < 8; ++r) {
        float* crow = C + (size_t)(rbase + r) * ldc;
#pragma unroll
        for (int t = 0; t < 4; ++t)
            crow[col0 + t * 16 + cr] = acc[t][r];
    }
}

// ---------------- RMSNorm + RoPE (per (s, head)) ----------------
__global__ __launch_bounds__(128) void rmsnorm_rope(
    const float* __restrict__ Qraw, const float* __restrict__ Kraw,
    const float* __restrict__ qnw, const float* __restrict__ knw,
    const float* __restrict__ cosp, const float* __restrict__ sinp,
    __bf16* __restrict__ qbf, __bf16* __restrict__ kbf)
{
    int s = blockIdx.x;
    int idx = blockIdx.y;
    int d = threadIdx.x;
    bool isQ = idx < NHEADS;
    int h = isQ ? idx : idx - NHEADS;
    const float* src = isQ ? (Qraw + (size_t)s * DMODEL + (size_t)h * HDIM)
                           : (Kraw + (size_t)s * (NKV * HDIM) + (size_t)h * HDIM);
    const float* w = isQ ? qnw : knw;
    __bf16* dst = isQ ? (qbf + (size_t)s * DMODEL + (size_t)h * HDIM)
                      : (kbf + (size_t)s * (NKV * HDIM) + (size_t)h * HDIM);
    float x = src[d];
    float sq = x * x;
#pragma unroll
    for (int off = 16; off; off >>= 1) sq += __shfl_xor(sq, off);
    __shared__ float red[4];
    __shared__ float vals[HDIM];
    int lane = d & 31, wv = d >> 5;
    if (lane == 0) red[wv] = sq;
    __syncthreads();
    float inv = rsqrtf((red[0] + red[1] + red[2] + red[3]) / (float)HDIM + 1e-6f);
    float val = w[d] * x * inv;
    vals[d] = val;
    __syncthreads();
    float other = (d < HDIM / 2) ? -vals[d + HDIM / 2] : vals[d - HDIM / 2];
    float c = cosp[(size_t)s * HDIM + d];
    float sn = sinp[(size_t)s * HDIM + d];
    dst[d] = f2bf(val * c + other * sn);
}

// ---------------- row softmax, in place over attn region of d_out ----------------
__global__ __launch_bounds__(256) void softmax_rows(float* __restrict__ attn)
{
    size_t row = blockIdx.x;
    float* p = attn + row * (size_t)S_LEN;
    __shared__ float buf[S_LEN];
    __shared__ float red[8];
    int tid = threadIdx.x, lane = tid & 31, wv = tid >> 5;
    float m = MASKV;
    for (int i = tid; i < S_LEN; i += 256) { float x = p[i]; buf[i] = x; m = fmaxf(m, x); }
#pragma unroll
    for (int off = 16; off; off >>= 1) m = fmaxf(m, __shfl_xor(m, off));
    if (lane == 0) red[wv] = m;
    __syncthreads();
    m = red[0];
#pragma unroll
    for (int i = 1; i < 8; ++i) m = fmaxf(m, red[i]);
    float sum = 0.f;
    for (int i = tid; i < S_LEN; i += 256) { float e = __expf(buf[i] - m); buf[i] = e; sum += e; }
#pragma unroll
    for (int off = 16; off; off >>= 1) sum += __shfl_xor(sum, off);
    __syncthreads();
    if (lane == 0) red[wv] = sum;
    __syncthreads();
    float tot = red[0] + red[1] + red[2] + red[3] + red[4] + red[5] + red[6] + red[7];
    float invs = 1.0f / tot;
    for (int i = tid; i < S_LEN; i += 256) p[i] = buf[i] * invs;
}

// ---------------- conversions ----------------
__global__ void f32_to_bf16(const float* __restrict__ in, __bf16* __restrict__ out, int n) {
    int i = blockIdx.x * blockDim.x + threadIdx.x;
    int stride = gridDim.x * blockDim.x;
    for (; i < n; i += stride) out[i] = f2bf(in[i]);
}

// V [S, KV*HD] fp32  ->  Vt [KV*HD, S] bf16
__global__ void v_transpose_bf16(const float* __restrict__ v, __bf16* __restrict__ vT) {
    const int total = S_LEN * NKV * HDIM;
    int i = blockIdx.x * blockDim.x + threadIdx.x;
    int stride = gridDim.x * blockDim.x;
    for (; i < total; i += stride) {
        int s = i / (NKV * HDIM);
        int e = i % (NKV * HDIM);
        vT[(size_t)e * S_LEN + s] = f2bf(v[i]);
    }
}

// ---------------- host orchestration ----------------
extern "C" void kernel_launch(void* const* d_in, const int* in_sizes, int n_in,
                              void* d_out, int out_size, void* d_ws, size_t ws_size,
                              hipStream_t stream) {
    (void)in_sizes; (void)n_in; (void)out_size; (void)ws_size;
    const float* hidden = (const float*)d_in[0];
    const float* cosp   = (const float*)d_in[1];
    const float* sinp   = (const float*)d_in[2];
    /* d_in[3] attention_mask: causal mask is computed analytically in-kernel */
    const float* q_w    = (const float*)d_in[4];
    const float* k_w    = (const float*)d_in[5];
    const float* v_w    = (const float*)d_in[6];
    const float* o_w    = (const float*)d_in[7];
    const float* qnw    = (const float*)d_in[8];
    const float* knw    = (const float*)d_in[9];

    float* out  = (float*)d_out;
    float* attn = out + (size_t)S_LEN * DMODEL;   // fp32 attn probs region of d_out

    char* ws = (char*)d_ws;
    size_t off = 0;
    auto alloc = [&](size_t bytes) -> void* {
        void* p = ws + off; off += (bytes + 255) & ~(size_t)255; return p;
    };
    const size_t SD = (size_t)S_LEN * DMODEL;       // 4M
    const size_t SK = (size_t)S_LEN * NKV * HDIM;   // 2M

    __bf16* hid_bf = (__bf16*)alloc(SD * 2);
    __bf16* qw_bf  = (__bf16*)alloc(SD * 2);        // (2048,2048)
    __bf16* kw_bf  = (__bf16*)alloc(SK * 2);        // (1024,2048) flat
    __bf16* vw_bf  = (__bf16*)alloc(SK * 2);
    __bf16* ow_bf  = (__bf16*)alloc(SD * 2);
    float*  Qraw   = (float*)alloc(SD * 4);
    float*  Kraw   = (float*)alloc(SK * 4);
    float*  Vraw   = (float*)alloc(SK * 4);
    __bf16* q_bf   = (__bf16*)alloc(SD * 2);
    __bf16* k_bf   = (__bf16*)alloc(SK * 2);
    __bf16* vT     = (__bf16*)alloc(SK * 2);
    float*  AO     = (float*)alloc(SD * 4);
    __bf16* ao_bf  = (__bf16*)alloc(SD * 2);

    const dim3 cvB(256), cvG(2048);
    f32_to_bf16<<<cvG, cvB, 0, stream>>>(hidden, hid_bf, (int)SD);
    f32_to_bf16<<<cvG, cvB, 0, stream>>>(q_w, qw_bf, (int)SD);
    f32_to_bf16<<<cvG, cvB, 0, stream>>>(k_w, kw_bf, (int)SK);
    f32_to_bf16<<<cvG, cvB, 0, stream>>>(v_w, vw_bf, (int)SK);
    f32_to_bf16<<<cvG, cvB, 0, stream>>>(o_w, ow_bf, (int)SD);

    // QKV projections: C[S,E] = hid * W^T
    gemm_bf16_nt<<<dim3(S_LEN/128, DMODEL/64, 1), 256, 0, stream>>>(
        hid_bf, qw_bf, Qraw, DMODEL, DMODEL, DMODEL, DMODEL, 0, 0, 1, 0, 1.0f, 0);
    gemm_bf16_nt<<<dim3(S_LEN/128, (NKV*HDIM)/64, 1), 256, 0, stream>>>(
        hid_bf, kw_bf, Kraw, DMODEL, DMODEL, DMODEL, NKV*HDIM, 0, 0, 1, 0, 1.0f, 0);
    gemm_bf16_nt<<<dim3(S_LEN/128, (NKV*HDIM)/64, 1), 256, 0, stream>>>(
        hid_bf, vw_bf, Vraw, DMODEL, DMODEL, DMODEL, NKV*HDIM, 0, 0, 1, 0, 1.0f, 0);

    rmsnorm_rope<<<dim3(S_LEN, NHEADS + NKV, 1), 128, 0, stream>>>(
        Qraw, Kraw, qnw, knw, cosp, sinp, q_bf, k_bf);
    v_transpose_bf16<<<cvG, cvB, 0, stream>>>(Vraw, vT);

    // scores: per head  attn[h] = scale * Q_h K_h^T + causal   (GQA: K head = h/2)
    gemm_bf16_nt<<<dim3(S_LEN/128, S_LEN/64, NHEADS), 256, 0, stream>>>(
        q_bf, k_bf, attn, HDIM, DMODEL, NKV*HDIM, S_LEN,
        /*strideA=*/HDIM, /*strideB=*/HDIM, /*repB=*/NREP,
        /*strideC=*/(long long)S_LEN * S_LEN,
        0.08838834764831845f, /*causal=*/1);

    softmax_rows<<<NHEADS * S_LEN, 256, 0, stream>>>(attn);

    // AO[:, h*128:(h+1)*128] = P_h * V_h   (B = V^T per kv head)
    gemm_f32a_nt<<<dim3(S_LEN/128, HDIM/64, NHEADS), 256, 0, stream>>>(
        attn, vT, AO, S_LEN, S_LEN, S_LEN, DMODEL,
        (long long)S_LEN * S_LEN, (long long)HDIM * S_LEN, NREP, (long long)HDIM);

    f32_to_bf16<<<cvG, cvB, 0, stream>>>(AO, ao_bf, (int)SD);

    // out = AO * o_w^T
    gemm_bf16_nt<<<dim3(S_LEN/128, DMODEL/64, 1), 256, 0, stream>>>(
        ao_bf, ow_bf, out, DMODEL, DMODEL, DMODEL, DMODEL, 0, 0, 1, 0, 1.0f, 0);
}
